// GNNClustering_73985106641234
// MI455X (gfx1250) — matrix operand order repaired and checked by reading, not compiled, not run on hardware
//
#include <hip/hip_runtime.h>
#include <hip/hip_bf16.h>
#include <stdint.h>

// ---------------------------------------------------------------------------
// 3-layer GCN for gfx1250 (MI455X).
//   GEMMs   : bf16 V_WMMA_F32_16X16X32_BF16. Each wave computes a 16x(NT*16)
//             output stripe: A-fragment loaded once per K-step, NT WMMAs
//             issued against it (4x less A traffic, 4 independent accums).
//   Edges   : float4 gathers + global_atomic_add_f32 scatters (L2-resident:
//             h/agg arrays are <=25.6MB << 192MB L2).
//   Norm    : deg via atomics once, dinv = rsqrt(deg), reused by all layers.
// ---------------------------------------------------------------------------

typedef __attribute__((ext_vector_type(16))) __bf16 v16bf;
typedef __attribute__((ext_vector_type(8)))  float  v8f;

// ---------------- degree / normalization ----------------
__global__ void k_deg_init(float* deg, int n) {
  int i = blockIdx.x * blockDim.x + threadIdx.x;
  if (i < n) deg[i] = 1.0f;  // self-loop
}

__global__ void k_deg_accum(const int* __restrict__ ei, float* deg, int E) {
  int i = blockIdx.x * blockDim.x + threadIdx.x;
  if (i < E) atomicAdd(&deg[ei[(size_t)E + i]], 1.0f);  // dst = ei[1][e]
}

__global__ void k_rsqrt_inplace(float* d, int n) {
  int i = blockIdx.x * blockDim.x + threadIdx.x;
  if (i < n) d[i] = rsqrtf(d[i]);
}

// ---------------- conversions ----------------
__global__ void k_f32_to_bf16(const float* __restrict__ in, __bf16* __restrict__ out,
                              long long n) {
  long long stride = (long long)gridDim.x * blockDim.x;
  for (long long i = (long long)blockIdx.x * blockDim.x + threadIdx.x; i < n; i += stride)
    out[i] = (__bf16)in[i];
}

// W [K][Nout] row-major f32  ->  Wt [Nout][K] row-major bf16 (so B-frag reads
// are two contiguous 16B runs per lane).
__global__ void k_weight_transpose(const float* __restrict__ W, __bf16* __restrict__ Wt,
                                   int K, int Nout) {
  int total = K * Nout;
  int i = blockIdx.x * blockDim.x + threadIdx.x;
  if (i < total) {
    int k = i / Nout;
    int n = i - k * Nout;
    Wt[(size_t)n * K + k] = (__bf16)W[i];
  }
}

__global__ void k_zero_f32(float* __restrict__ p, long long n) {
  long long stride = (long long)gridDim.x * blockDim.x;
  for (long long i = (long long)blockIdx.x * blockDim.x + threadIdx.x; i < n; i += stride)
    p[i] = 0.0f;
}

// ---------------- WMMA GEMM: H[M,Nout] = A[M,K](bf16) @ Wt^T ----------------
// A row-major bf16 [M][K]; Bt row-major bf16 [Nout][K]; H f32 [M][Nout].
// One wave computes a 16 x (NT*16) stripe; K stepped by 32 per WMMA.
// Requires: M%16==0, Nout%(16*NT)==0, K%32==0 (true for all three layers).
template <int NT>
__global__ void k_gemm_wmma_bf16(const __bf16* __restrict__ A,
                                 const __bf16* __restrict__ Bt,
                                 float* __restrict__ H,
                                 int M, int K, int Nout) {
  const int lane   = threadIdx.x & 31;
  const int wid    = blockIdx.x * (blockDim.x >> 5) + (threadIdx.x >> 5);
  const int groupsN = Nout / (16 * NT);
  const int ngroups = (M >> 4) * groupsN;
  if (wid >= ngroups) return;  // wave-uniform: EXEC stays all-ones for WMMA

  const int tm   = wid / groupsN;
  const int tg   = wid - tm * groupsN;   // N-group index
  const int half = lane >> 4;            // 0: lanes 0-15, 1: lanes 16-31
  const int l16  = lane & 15;

  const __bf16* Ap = A + (size_t)(tm * 16 + l16) * K;  // row M = l16 (both halves)
  const __bf16* Bp[NT];
#pragma unroll
  for (int t = 0; t < NT; ++t)           // col N = l16 within tile t
    Bp[t] = Bt + (size_t)(tg * 16 * NT + t * 16 + l16) * K;

  v8f c[NT];
#pragma unroll
  for (int t = 0; t < NT; ++t) c[t] = (v8f){};

  for (int kk = 0; kk < K; kk += 32) {
    // A 16x32 bf16 layout: VGPR v holds K = 16*(v/4) + 8*half + 2*(v%4) (+1)
    v16bf a;
#pragma unroll
    for (int v = 0; v < 8; ++v) {
      int ka = kk + ((v >> 2) << 4) + (half << 3) + ((v & 3) << 1);
      a[2 * v]     = Ap[ka];
      a[2 * v + 1] = Ap[ka + 1];
    }
    // B 32x16 bf16 layout: VGPR v holds K = 16*(v/4) + 8*half + 2*(v%4) (+1), lane = N
#pragma unroll
    for (int t = 0; t < NT; ++t) {
      v16bf b;
#pragma unroll
      for (int v = 0; v < 8; ++v) {
        int kb = kk + ((v >> 2) << 4) + (half << 3) + ((v & 3) << 1);
        b[2 * v]     = Bp[t][kb];
        b[2 * v + 1] = Bp[t][kb + 1];
      }
      c[t] = __builtin_amdgcn_wmma_f32_16x16x32_bf16(
          /*neg_a=*/false, a, /*neg_b=*/false, b,
          /*c_mod=*/(short)0, c[t], /*reuse_a=*/false, /*reuse_b=*/false);
    }
  }

  // C/D f32 16x16: lane half selects M base (0 or 8), VGPR v = M offset, lane%16 = N
#pragma unroll
  for (int t = 0; t < NT; ++t) {
    float* Hp = H + (size_t)(tm * 16 + half * 8) * Nout + tg * 16 * NT + t * 16 + l16;
#pragma unroll
    for (int v = 0; v < 8; ++v)
      Hp[(size_t)v * Nout] = c[t][v];
  }
}

// ---------------- edge aggregation: agg[dst] += h[src] * coef ----------------
// One thread handles a float4 chunk of one edge's feature row. F4 = F/4.
__global__ void k_aggregate(const float* __restrict__ h, const int* __restrict__ ei,
                            const float* __restrict__ dinv, float* __restrict__ agg,
                            int E, int F4) {
  long long total  = (long long)E * F4;
  long long stride = (long long)gridDim.x * blockDim.x;
  for (long long i = (long long)blockIdx.x * blockDim.x + threadIdx.x; i < total;
       i += stride) {
    int e = (int)(i / F4);
    int c = (int)(i - (long long)e * F4);
    int s = ei[e];                 // src
    int d = ei[(size_t)E + e];     // dst
    float coef = dinv[s] * dinv[d];
    const float4 hv = *(const float4*)(h + ((size_t)s * F4 + c) * 4);
    float* out = agg + ((size_t)d * F4 + c) * 4;
    atomicAdd(out + 0, hv.x * coef);
    atomicAdd(out + 1, hv.y * coef);
    atomicAdd(out + 2, hv.z * coef);
    atomicAdd(out + 3, hv.w * coef);
  }
}

// ---------------- finalize: out = agg + h*dinv^2 + b, optional ReLU ----------
__global__ void k_finalize(const float* __restrict__ agg, const float* __restrict__ h,
                           const float* __restrict__ dinv, const float* __restrict__ bias,
                           float* __restrict__ out_f32, __bf16* __restrict__ out_bf16,
                           int N, int F, int relu) {
  long long total  = (long long)N * F;
  long long stride = (long long)gridDim.x * blockDim.x;
  for (long long i = (long long)blockIdx.x * blockDim.x + threadIdx.x; i < total;
       i += stride) {
    int n = (int)(i / F);
    int f = (int)(i - (long long)n * F);
    float di = dinv[n];
    float v = agg[i] + h[i] * di * di + bias[f];
    if (relu) v = fmaxf(v, 0.0f);
    if (out_f32)  out_f32[i]  = v;
    if (out_bf16) out_bf16[i] = (__bf16)v;
  }
}

// ---------------------------------------------------------------------------
static inline size_t align256(size_t x) { return (x + 255) & ~(size_t)255; }

extern "C" void kernel_launch(void* const* d_in, const int* in_sizes, int n_in,
                              void* d_out, int out_size, void* d_ws, size_t ws_size,
                              hipStream_t stream) {
  const float* x  = (const float*)d_in[0];
  const int*   ei = (const int*)d_in[1];
  const float* W1 = (const float*)d_in[2];
  const float* b1 = (const float*)d_in[3];
  const float* W2 = (const float*)d_in[4];
  const float* b2 = (const float*)d_in[5];
  const float* W3 = (const float*)d_in[6];
  const float* b3 = (const float*)d_in[7];
  float* out = (float*)d_out;

  const int N = in_sizes[0] / 128;   // 50000
  const int E = in_sizes[1] / 2;     // 600000

  // ---- workspace carve-up ----
  char* ws = (char*)d_ws;
  size_t off = 0;
  float*  dinv = (float*)(ws + off);  off += align256((size_t)N * 4);
  __bf16* xb   = (__bf16*)(ws + off); off += align256((size_t)N * 128 * 2);
  float*  h    = (float*)(ws + off);  off += align256((size_t)N * 128 * 4);
  float*  agg  = (float*)(ws + off);  off += align256((size_t)N * 128 * 4);
  __bf16* w1t  = (__bf16*)(ws + off); off += align256((size_t)128 * 128 * 2);
  __bf16* w2t  = (__bf16*)(ws + off); off += align256((size_t)64 * 128 * 2);
  __bf16* w3t  = (__bf16*)(ws + off); off += align256((size_t)16 * 64 * 2);
  (void)ws_size; (void)n_in; (void)out_size;

  const int TB = 256;  // 8 wave32 per block
  auto blocks = [&](long long work) { return (int)((work + TB - 1) / TB); };

  // ---- normalization (once, reused by all layers) ----
  k_deg_init<<<blocks(N), TB, 0, stream>>>(dinv, N);
  k_deg_accum<<<blocks(E), TB, 0, stream>>>(ei, dinv, E);
  k_rsqrt_inplace<<<blocks(N), TB, 0, stream>>>(dinv, N);

  // ---- bf16 conversions ----
  k_f32_to_bf16<<<blocks((long long)N * 128), TB, 0, stream>>>(x, xb, (long long)N * 128);
  k_weight_transpose<<<blocks(128 * 128), TB, 0, stream>>>(W1, w1t, 128, 128);
  k_weight_transpose<<<blocks(128 * 64),  TB, 0, stream>>>(W2, w2t, 128, 64);
  k_weight_transpose<<<blocks(64 * 16),   TB, 0, stream>>>(W3, w3t, 64, 16);

  // ---- layer 1: 128 -> 128, ReLU ----
  {
    const int K = 128, F = 128;
    int ngroups = (N / 16) * (F / 64);  // NT=4
    k_gemm_wmma_bf16<4><<<(ngroups + 7) / 8, TB, 0, stream>>>(xb, w1t, h, N, K, F);
    k_zero_f32<<<blocks((long long)N * F), TB, 0, stream>>>(agg, (long long)N * F);
    k_aggregate<<<blocks((long long)E * (F / 4)), TB, 0, stream>>>(h, ei, dinv, agg, E, F / 4);
    k_finalize<<<blocks((long long)N * F), TB, 0, stream>>>(agg, h, dinv, b1,
                                                            nullptr, xb, N, F, 1);
  }

  // ---- layer 2: 128 -> 64, ReLU ----
  {
    const int K = 128, F = 64;
    int ngroups = (N / 16) * (F / 64);  // NT=4
    k_gemm_wmma_bf16<4><<<(ngroups + 7) / 8, TB, 0, stream>>>(xb, w2t, h, N, K, F);
    k_zero_f32<<<blocks((long long)N * F), TB, 0, stream>>>(agg, (long long)N * F);
    k_aggregate<<<blocks((long long)E * (F / 4)), TB, 0, stream>>>(h, ei, dinv, agg, E, F / 4);
    k_finalize<<<blocks((long long)N * F), TB, 0, stream>>>(agg, h, dinv, b2,
                                                            nullptr, xb, N, F, 1);
  }

  // ---- layer 3: 64 -> 16, no ReLU, f32 to d_out ----
  {
    const int K = 64, F = 16;
    int ngroups = (N / 16) * (F / 16);  // NT=1
    k_gemm_wmma_bf16<1><<<(ngroups + 7) / 8, TB, 0, stream>>>(xb, w3t, h, N, K, F);
    k_zero_f32<<<blocks((long long)N * F), TB, 0, stream>>>(agg, (long long)N * F);
    k_aggregate<<<blocks((long long)E * (F / 4)), TB, 0, stream>>>(h, ei, dinv, agg, E, F / 4);
    k_finalize<<<blocks((long long)N * F), TB, 0, stream>>>(agg, h, dinv, b3,
                                                            out, nullptr, N, F, 0);
  }
}
